// DirectEncodingModel_1460288881281
// MI455X (gfx1250) — matrix operand order repaired and checked by reading, not compiled, not run on hardware
//
#include <hip/hip_runtime.h>

typedef __attribute__((ext_vector_type(16))) _Float16 v16h;
typedef __attribute__((ext_vector_type(8)))  _Float16 v8h;
typedef __attribute__((ext_vector_type(2)))  _Float16 v2h;
typedef __attribute__((ext_vector_type(8)))  float    v8f;

#define B_TOTAL 131072
#define DIM     256
#define NLEV    3
#define NGRP    4
#define KDIM    128
#define UDIM    64
#define OUT_N   10
#define ROWS    32
#define GC      (NGRP*KDIM)          // 512 gathered columns per level
#define WB_LVL  (NLEV*NGRP*KDIM*UDIM)   // 98304 halves of level B-fragments
#define WB_OUT  (8*32*16)               // 4096 halves of projection B-fragments

// CDNA5 hardware tanh (TRANS op). Builtin if declared, else inline asm with a
// v_nop to satisfy the trans-op result hazard rule (ISA 7.4).
__device__ inline float hw_tanh(float x) {
#if __has_builtin(__builtin_amdgcn_tanhf)
    return __builtin_amdgcn_tanhf(x);
#else
    float r;
    asm volatile("v_tanh_f32 %0, %1\n\tv_nop" : "=v"(r) : "v"(x));
    return r;
#endif
}

// ---------------------------------------------------------------------------
// Prep: swizzle weights (f32) into f16 WMMA B-fragment layout in workspace.
//   Levels:     W[L][G][K][U]  -> fragments (l,g,kc,nt), 32 lanes x 16 halves
//   Projection: W_out[D][10]   -> fragments (kc 0..7), N padded to 16 w/ zeros
// B layout (32x16 B of 16x16x32 f16 WMMA):
//   lane 0-15 : n = lane,    halves 0..7 = K 0..7,  halves 8..15 = K 16..23
//   lane 16-31: n = lane-16, halves 0..7 = K 8..15, halves 8..15 = K 24..31
// ---------------------------------------------------------------------------
__global__ void prep_wb(const float* __restrict__ W,
                        const float* __restrict__ W_out,
                        _Float16* __restrict__ Wb) {
    int t = blockIdx.x * blockDim.x + threadIdx.x;
    if (t < WB_LVL) {
        int h    =  t        & 15;
        int lane = (t >> 4)  & 31;
        int nt   = (t >> 9)  & 3;
        int kc   = (t >> 11) & 3;
        int g    = (t >> 13) & 3;
        int l    =  t >> 15;
        int k = kc * 32 + ((h < 8) ? h : h + 8) + ((lane >= 16) ? 8 : 0);
        int n = nt * 16 + (lane & 15);
        Wb[t] = (_Float16)W[(((l * NGRP + g) * KDIM) + k) * UDIM + n];
    } else if (t < WB_LVL + WB_OUT) {
        int t2   = t - WB_LVL;
        int h    =  t2       & 15;
        int lane = (t2 >> 4) & 31;
        int kc   =  t2 >> 9;                 // 0..7
        int d = kc * 32 + ((h < 8) ? h : h + 8) + ((lane >= 16) ? 8 : 0);
        int n = lane & 15;
        Wb[t] = (n < OUT_N) ? (_Float16)W_out[d * OUT_N + n] : (_Float16)0.0f;
    }
}

// ---------------------------------------------------------------------------
// Fused: 3x (gather -> grouped WMMA GEMM -> hw tanh) + WMMA projection.
// h stays resident in LDS; x is read from HBM exactly once.
// ---------------------------------------------------------------------------
__global__ __launch_bounds__(256) void fused_mlp(
    const float*    __restrict__ x,
    const int*      __restrict__ idx,
    const float*    __restrict__ bias,
    const _Float16* __restrict__ Wb,
    const float*    __restrict__ b_out,
    float*          __restrict__ out)
{
    __shared__ _Float16 h_lds[ROWS][DIM];          // 16 KB
    __shared__ _Float16 gbuf [ROWS][GC];           // 32 KB
    __shared__ int      idxb [GC];                 //  2 KB
    __shared__ float    ppart[2][4][16][16];       //  8 KB projection partials

    const int t    = threadIdx.x;
    const int lane = t & 31;
    const int wave = t >> 5;
    const int mt   = wave >> 2;       // 16-row tile this wave owns (levels)
    const int g    = wave & 3;        // group this wave owns (levels)
    const int hi   = (lane >= 16) ? 1 : 0;
    const int mrow = mt * 16 + (lane & 15);
    const int row0 = blockIdx.x * ROWS;

    // Stage x (f32 -> f16) into LDS.
    #pragma unroll
    for (int i = 0; i < (ROWS * DIM) / 256; ++i) {
        int e = i * 256 + t;
        int r = e >> 8, c = e & 255;
        h_lds[r][c] = (_Float16)x[(size_t)(row0 + r) * DIM + c];
    }

    for (int l = 0; l < NLEV; ++l) {
        __syncthreads();                           // h_lds ready / prev reads done
        idxb[t]       = idx[l * GC + t];
        idxb[t + 256] = idx[l * GC + t + 256];
        __syncthreads();

        // Cooperative gather, two columns per thread -> one b32 store.
        #pragma unroll
        for (int i = 0; i < (ROWS * GC) / (256 * 2); ++i) {
            int e  = i * 256 + t;
            int r  = e >> 8;
            int c2 = (e & 255) * 2;
            v2h p;
            p[0] = h_lds[r][idxb[c2]];
            p[1] = h_lds[r][idxb[c2 + 1]];
            *(v2h*)&gbuf[r][c2] = p;
        }
        __syncthreads();                           // gbuf ready, h_lds writable

        // Wave: rows [mt*16,+16), group g, all 64 outputs of that group.
        v8f acc[4];
        #pragma unroll
        for (int nt = 0; nt < 4; ++nt) acc[nt] = (v8f){};

        #pragma unroll
        for (int kc = 0; kc < 4; ++kc) {
            int cb = g * KDIM + kc * 32 + hi * 8;  // A: ISA 16-bit 16x32 layout
            v8h lo = *(const v8h*)&gbuf[mrow][cb];
            v8h hh = *(const v8h*)&gbuf[mrow][cb + 16];
            v16h a;
            #pragma unroll
            for (int j = 0; j < 8; ++j) { a[j] = lo[j]; a[j + 8] = hh[j]; }

            const _Float16* wb =
                Wb + ((size_t)((((l * NGRP + g) * 4 + kc) * 4) * 32 + lane)) * 16;
            #pragma unroll
            for (int nt = 0; nt < 4; ++nt) {
                v16h bf = *(const v16h*)(wb + (size_t)nt * (32 * 16));
                acc[nt] = __builtin_amdgcn_wmma_f32_16x16x32_f16(
                    false, a, false, bf, (short)0, acc[nt], false, false);
            }
        }

        // bias + hardware tanh, write new h (C layout: VGPR r -> M = r + hi*8).
        #pragma unroll
        for (int nt = 0; nt < 4; ++nt) {
            int u    = nt * 16 + (lane & 15);
            float bv = bias[(l * NGRP + g) * UDIM + u];
            int col  = g * UDIM + u;
            #pragma unroll
            for (int r = 0; r < 8; ++r) {
                int rr = mt * 16 + r + hi * 8;
                h_lds[rr][col] = (_Float16)hw_tanh(acc[nt][r] + bv);
            }
        }
    }

    __syncthreads();
    // Projection out = h3 @ W_out (+b_out) via WMMA, OUT padded 10->16.
    // Wave w: M-tile (w&1), k-chunks {2*(w>>1), 2*(w>>1)+1}; partials to LDS.
    {
        const int pmt  = wave & 1;
        const int kg   = wave >> 1;               // 0..3
        const int prow = pmt * 16 + (lane & 15);
        v8f pacc = (v8f){};
        #pragma unroll
        for (int kk = 0; kk < 2; ++kk) {
            int kc = kg * 2 + kk;
            int cb = kc * 32 + hi * 8;
            v8h lo = *(const v8h*)&h_lds[prow][cb];
            v8h hh = *(const v8h*)&h_lds[prow][cb + 16];
            v16h a;
            #pragma unroll
            for (int j = 0; j < 8; ++j) { a[j] = lo[j]; a[j + 8] = hh[j]; }
            v16h bf = *(const v16h*)(Wb + WB_LVL + ((size_t)kc * 32 + lane) * 16);
            pacc = __builtin_amdgcn_wmma_f32_16x16x32_f16(
                false, a, false, bf, (short)0, pacc, false, false);
        }
        #pragma unroll
        for (int r = 0; r < 8; ++r)
            ppart[pmt][kg][r + hi * 8][lane & 15] = pacc[r];
    }
    __syncthreads();

    // 4-way partial reduction: all 256 threads, one (row, out) each (pairs 0..255 only).
    if (t < ROWS * OUT_N) {
        int r = t / OUT_N, o = t % OUT_N;
        float s = b_out[o];
        #pragma unroll
        for (int kg = 0; kg < 4; ++kg)
            s += ppart[r >> 4][kg][r & 15][o];
        out[(size_t)(row0 + r) * OUT_N + o] = s;
    }
}

extern "C" void kernel_launch(void* const* d_in, const int* in_sizes, int n_in,
                              void* d_out, int out_size, void* d_ws, size_t ws_size,
                              hipStream_t stream) {
    const float* x     = (const float*)d_in[0];
    const int*   idx   = (const int*)  d_in[1];
    const float* W     = (const float*)d_in[2];
    const float* b     = (const float*)d_in[3];
    const float* W_out = (const float*)d_in[4];
    const float* b_out = (const float*)d_in[5];
    float*       out   = (float*)d_out;
    _Float16*    Wb    = (_Float16*)d_ws;   // 200 KB of f16 B-fragments

    int prep_elems = WB_LVL + WB_OUT;       // 102400
    prep_wb<<<(prep_elems + 255) / 256, 256, 0, stream>>>(W, W_out, Wb);
    fused_mlp<<<B_TOTAL / ROWS, 256, 0, stream>>>(x, idx, b, Wb, b_out, out);
}
